// CrossAttention_43258910605402
// MI455X (gfx1250) — matrix operand order repaired and checked by reading, not compiled, not run on hardware
//
#include <hip/hip_runtime.h>
#include <stdint.h>

// ---------------- problem constants ----------------
#define NQ    6912   // query tokens
#define NK    3072   // key tokens
#define QD    320    // model dim
#define INNER 512    // heads*dim_head
#define HEADS 8
#define DHEAD 64

#if defined(__AMDGCN__) && __has_builtin(__builtin_amdgcn_tensor_load_to_lds)
#define HAVE_TDM 1
#else
#define HAVE_TDM 0
#endif

// ---------------- vector / fragment types ----------------
typedef __attribute__((ext_vector_type(16))) __bf16    v16bf;
typedef __attribute__((ext_vector_type(8)))  float     v8f;
typedef __attribute__((ext_vector_type(4)))  unsigned  u32x4;
typedef __attribute__((ext_vector_type(4)))  float     f32x4;
typedef __attribute__((ext_vector_type(8)))  int       i32x8;
typedef __attribute__((ext_vector_type(4)))  int       i32x4;

union FragAB { unsigned u[8]; u32x4 q[2]; v16bf v; };
union FragC  { float f[8]; v8f v; };

static __device__ __forceinline__ unsigned short f2bf(float x) {
  unsigned u = __float_as_uint(x);
  u += 0x7FFFu + ((u >> 16) & 1u);        // round-to-nearest-even
  return (unsigned short)(u >> 16);
}

static __device__ __forceinline__ v8f wmma_bf16(v16bf a, v16bf b, v8f c) {
  return __builtin_amdgcn_wmma_f32_16x16x32_bf16(false, a, false, b,
                                                 (short)0, c, false, false);
}

// 16-lane butterfly reductions via DPP ROW_XMASK (no LDS traffic).
template <int XM>
static __device__ __forceinline__ float dpp_xor(float x) {
  return __int_as_float(__builtin_amdgcn_update_dpp(
      0, __float_as_int(x), 0x160 | XM, 0xF, 0xF, true));
}
static __device__ __forceinline__ float redmax16(float x) {
  x = fmaxf(x, dpp_xor<1>(x));
  x = fmaxf(x, dpp_xor<2>(x));
  x = fmaxf(x, dpp_xor<4>(x));
  x = fmaxf(x, dpp_xor<8>(x));
  return x;
}
static __device__ __forceinline__ float redsum16(float x) {
  x += dpp_xor<1>(x);
  x += dpp_xor<2>(x);
  x += dpp_xor<4>(x);
  x += dpp_xor<8>(x);
  return x;
}

// Async global -> LDS (128b per lane), tracked by ASYNCcnt.
static __device__ __forceinline__ void async_g2l_b128(unsigned lds_off,
                                                      const void* gaddr) {
  asm volatile("global_load_async_to_lds_b128 %0, %1, off"
               :: "v"(lds_off), "v"(gaddr) : "memory");
}
static __device__ __forceinline__ void wait_async0() {
  asm volatile("s_wait_asynccnt 0x0" ::: "memory");
}

#if HAVE_TDM
// TDM: DMA a 32-row x 64-col bf16 tile (row stride 512 elems) into LDS.
// D# built per ISA ch.8: group0 = {count=1, lds_addr, global_addr, type=2},
// group1 = {data_size=2B, tensor 64x32, tile 64x32, dim0_stride=512}.
// This toolchain exposes the 6-arg builtin (extra int32x8 before cpol).
static __device__ __forceinline__ void tdm_load_k(const unsigned short* gtile,
                                                  unsigned lds_byte) {
  unsigned long long ga = (unsigned long long)(size_t)gtile;
  u32x4 g0 = {1u, lds_byte, (unsigned)ga,
              (unsigned)((ga >> 32) & 0x01FFFFFFull) | 0x80000000u};
  i32x8 g1 = {0x00010000, (64 << 16), (32 << 16), (64 << 16), 32, 512, 0, 0};
  i32x4 gz4 = {0, 0, 0, 0};
  i32x8 gz8 = {0, 0, 0, 0, 0, 0, 0, 0};
  __builtin_amdgcn_tensor_load_to_lds(g0, g1, gz4, gz4, gz8, 0);
}
#endif

// ---------------- mask factorization ----------------
__global__ void build_masks(const float* __restrict__ m1,
                            const float* __restrict__ m2,
                            unsigned char* __restrict__ m1q,
                            unsigned* __restrict__ m2kbits) {
  int i = blockIdx.x * blockDim.x + threadIdx.x;
  if (i < NQ) {                       // 96 x 72 target from 512 x 384
    int r = i / 72, c = i % 72;
    int hi = (r * 512) / 96, wi = (c * 384) / 72;
    m1q[i] = (m1[hi * 384 + wi] >= 0.5f) ? 1 : 0;
  }
  if (i < NK / 32) {                  // key mask packed 32 bits per word
    unsigned wbits = 0;
    for (int b = 0; b < 32; b++) {
      int j = i * 32 + b;             // 64 x 48 target from 512 x 384
      int r = j / 48, c = j % 48;
      wbits |= (m2[(r * 8) * 384 + c * 8] >= 0.5f ? 1u : 0u) << b;
    }
    m2kbits[i] = wbits;
  }
}

// ---------------- generic WMMA GEMM: C[M,N] = cscale*(A[M,K]*B[K,N]) (+bias)
template <bool A_BF16, bool OUT_F32_BIAS>
__global__ __launch_bounds__(256) void gemm_wmma(
    const void* __restrict__ Aptr, const float* __restrict__ B,
    const float* __restrict__ bias, void* __restrict__ Cptr,
    int M, int N, int K, float cscale) {
  __shared__ unsigned short As[128][32];   // row-major bf16 A tile
  __shared__ unsigned short BsT[64][32];   // B tile transposed (n x k)

  const int t = threadIdx.x;
  const int lane = t & 31;
  const int w = t >> 5;
  const int hsel = lane >> 4;
  const int col = lane & 15;
  const int r0 = blockIdx.x * 128;
  const int n0 = blockIdx.y * 64;

  FragC acc[4];
  for (int i = 0; i < 4; i++)
    for (int j = 0; j < 8; j++) acc[i].f[j] = 0.f;

  for (int k0 = 0; k0 < K; k0 += 32) {
    if (A_BF16) {   // async copy, bf16 already
      const unsigned short* A = (const unsigned short*)Aptr;
      int row = t >> 2, c = (t & 3) * 8;
      #pragma unroll
      for (int p = 0; p < 2; p++) {
        async_g2l_b128((unsigned)(size_t)&As[row + p * 64][c],
                       A + (size_t)(r0 + row + p * 64) * K + k0 + c);
      }
    } else {        // fp32 -> bf16 conversion through VGPRs
      const float* A = (const float*)Aptr;
      int row = t >> 3, c = (t & 7) * 4;
      #pragma unroll
      for (int p = 0; p < 4; p++) {
        f32x4 v = *(const f32x4*)(A + (size_t)(r0 + row + p * 32) * K + k0 + c);
        unsigned short* d = &As[row + p * 32][c];
        d[0] = f2bf(v.x); d[1] = f2bf(v.y); d[2] = f2bf(v.z); d[3] = f2bf(v.w);
      }
    }
    {
      int rr = t >> 6, cc = t & 63;
      #pragma unroll
      for (int p = 0; p < 8; p++) {
        int kk = rr + p * 4;
        BsT[cc][kk] = f2bf(B[(size_t)(k0 + kk) * N + n0 + cc]);
      }
    }
    if (A_BF16) wait_async0();
    __syncthreads();

    FragAB a;   // A-layout halves are contiguous 16B runs
    {
      const unsigned short* ap = &As[w * 16 + col][0];
      a.q[0] = *(const u32x4*)(ap + hsel * 16);
      a.q[1] = *(const u32x4*)(ap + hsel * 16 + 8);
    }
    #pragma unroll
    for (int tt = 0; tt < 4; tt++) {
      FragAB b;
      const unsigned short* src = &BsT[tt * 16 + col][hsel * 16];
      b.q[0] = *(const u32x4*)(src);
      b.q[1] = *(const u32x4*)(src + 8);
      acc[tt].v = wmma_bf16(a.v, b.v, acc[tt].v);
    }
    __syncthreads();
  }

  #pragma unroll
  for (int tt = 0; tt < 4; tt++) {
    int cc = n0 + tt * 16 + col;
    if (OUT_F32_BIAS) {
      float* Cf = (float*)Cptr;
      float bv = bias[cc];
      #pragma unroll
      for (int r = 0; r < 8; r++) {
        int rr = r0 + w * 16 + hsel * 8 + r;
        Cf[(size_t)rr * N + cc] = acc[tt].f[r] * cscale + bv;
      }
    } else {
      unsigned short* Cb = (unsigned short*)Cptr;
      #pragma unroll
      for (int r = 0; r < 8; r++) {
        int rr = r0 + w * 16 + hsel * 8 + r;
        Cb[(size_t)rr * N + cc] = f2bf(acc[tt].f[r] * cscale);
      }
    }
  }
}

// ---------------- flash attention ----------------
// 8 waves x 32 query rows; 32 keys/iter; double-buffered K (TDM) + V tiles.
// Key order inside a block is interleaved (K' = 2*(k%16) + k/16) so the
// P-bounce packs two bf16 per ds_store_b32; softmax is order-invariant.
__global__ __launch_bounds__(256) void flash_attn(
    const unsigned short* __restrict__ qb, const unsigned short* __restrict__ kb,
    const unsigned short* __restrict__ vb, const unsigned char* __restrict__ m1q,
    const unsigned* __restrict__ m2kb, unsigned short* __restrict__ aob) {
  __shared__ unsigned short Kt[2][32][64];   // key-rel x d  (TDM/async filled)
  __shared__ unsigned short VtT[2][64][32];  // d x key' (transposed stage)
  __shared__ unsigned short Pb[8][16][32];   // per-wave P bounce (C->A)

  const int t = threadIdx.x, lane = t & 31, w = t >> 5;
  const int hsel = lane >> 4, col = lane & 15;
  const int h = blockIdx.y;
  const int qbase = blockIdx.x * 256 + w * 32;
  const int krel = t >> 3, c8 = (t & 7) * 8;
  const int kprm = ((krel & 15) << 1) | (krel >> 4);   // interleaved key slot
  const size_t gvoff = (size_t)h * DHEAD + c8;

  // Q fragments for both row-blocks (1/sqrt(d) folded in upstream).
  FragAB qa[2][2];
  #pragma unroll
  for (int rb = 0; rb < 2; rb++) {
    const unsigned short* qrow =
        qb + (size_t)(qbase + rb * 16 + col) * INNER + h * DHEAD;
    qa[rb][0].q[0] = *(const u32x4*)(qrow + hsel * 16);
    qa[rb][0].q[1] = *(const u32x4*)(qrow + hsel * 16 + 8);
    qa[rb][1].q[0] = *(const u32x4*)(qrow + 32 + hsel * 16);
    qa[rb][1].q[1] = *(const u32x4*)(qrow + 40 + hsel * 16);
  }

  float rmf[2][8];   // row gates {0,1}: masked score = fma(rmf, colbias, s)
  #pragma unroll
  for (int rb = 0; rb < 2; rb++)
    #pragma unroll
    for (int r = 0; r < 8; r++)
      rmf[rb][r] = m1q[qbase + rb * 16 + hsel * 8 + r] ? 1.0f : 0.0f;

  FragC o[2][4];
  for (int rb = 0; rb < 2; rb++)
    for (int i = 0; i < 4; i++)
      for (int j = 0; j < 8; j++) o[rb][i].f[j] = 0.f;
  float mrow[2][8], lrow[2][8];
  #pragma unroll
  for (int rb = 0; rb < 2; rb++)
    #pragma unroll
    for (int r = 0; r < 8; r++) { mrow[rb][r] = -3.0e38f; lrow[rb][r] = 0.f; }

  // ---- prologue: stage tile 0 into buffer 0 ----
  {
#if HAVE_TDM
    if (w == 0) tdm_load_k(kb + (size_t)h * DHEAD, (unsigned)(size_t)&Kt[0][0][0]);
#else
    async_g2l_b128((unsigned)(size_t)&Kt[0][krel][c8],
                   kb + (size_t)krel * INNER + gvoff);
#endif
    u32x4 vv = *(const u32x4*)(vb + (size_t)krel * INNER + gvoff);
    const unsigned short* ve = (const unsigned short*)&vv;
    #pragma unroll
    for (int e = 0; e < 8; e++) VtT[0][c8 + e][kprm] = ve[e];
#if HAVE_TDM
    if (w == 0) __builtin_amdgcn_s_wait_tensorcnt(0);
#else
    wait_async0();
#endif
  }
  __syncthreads();

  const int NT = NK / 32;
  for (int it = 0; it < NT; ++it) {
    const int cur = it & 1, nxt = cur ^ 1;
    const int j0 = it * 32;
    const bool more = (it + 1 < NT);

    // ---- kick off next tile's staging (latency hidden by compute) ----
    u32x4 vvn = {};
    if (more) {
#if HAVE_TDM
      if (w == 0) tdm_load_k(kb + (size_t)(j0 + 32) * INNER + h * DHEAD,
                             (unsigned)(size_t)&Kt[nxt][0][0]);
#else
      async_g2l_b128((unsigned)(size_t)&Kt[nxt][krel][c8],
                     kb + (size_t)(j0 + 32 + krel) * INNER + gvoff);
#endif
      vvn = *(const u32x4*)(vb + (size_t)(j0 + 32 + krel) * INNER + gvoff);
    }

    // ---- K fragments (2 key-subtiles x 2 d-halves), shared by both rb ----
    FragAB kf[2][2];
    {
      const unsigned short* kr0 = &Kt[cur][col][hsel * 16];
      const unsigned short* kr1 = &Kt[cur][16 + col][hsel * 16];
      kf[0][0].q[0] = *(const u32x4*)(kr0);       kf[0][0].q[1] = *(const u32x4*)(kr0 + 8);
      kf[0][1].q[0] = *(const u32x4*)(kr0 + 32);  kf[0][1].q[1] = *(const u32x4*)(kr0 + 40);
      kf[1][0].q[0] = *(const u32x4*)(kr1);       kf[1][0].q[1] = *(const u32x4*)(kr1 + 8);
      kf[1][1].q[0] = *(const u32x4*)(kr1 + 32);  kf[1][1].q[1] = *(const u32x4*)(kr1 + 40);
    }

    // ---- scores for both row-blocks ----
    FragC s[2][2];
    #pragma unroll
    for (int rb = 0; rb < 2; rb++) {
      for (int j = 0; j < 8; j++) { s[rb][0].f[j] = 0.f; s[rb][1].f[j] = 0.f; }
      s[rb][0].v = wmma_bf16(qa[rb][0].v, kf[0][0].v, s[rb][0].v);
      s[rb][0].v = wmma_bf16(qa[rb][1].v, kf[0][1].v, s[rb][0].v);
      s[rb][1].v = wmma_bf16(qa[rb][0].v, kf[1][0].v, s[rb][1].v);
      s[rb][1].v = wmma_bf16(qa[rb][1].v, kf[1][1].v, s[rb][1].v);
    }

    // ---- V fragments (interleaved key order, shared by both rb) ----
    FragAB vf[4];
    #pragma unroll
    for (int tt = 0; tt < 4; tt++) {
      const unsigned short* vp = &VtT[cur][tt * 16 + col][hsel * 16];
      vf[tt].q[0] = *(const u32x4*)vp;
      vf[tt].q[1] = *(const u32x4*)(vp + 8);
    }

    const unsigned mbits = m2kb[j0 >> 5];          // uniform scalar load
    const float cb0 = ((mbits >> col) & 1u)        ? -1e30f : 0.f;
    const float cb1 = ((mbits >> (col + 16)) & 1u) ? -1e30f : 0.f;

    #pragma unroll
    for (int rb = 0; rb < 2; rb++) {
      #pragma unroll
      for (int r = 0; r < 8; r++) {
        s[rb][0].f[r] = fmaf(rmf[rb][r], cb0, s[rb][0].f[r]);
        s[rb][1].f[r] = fmaf(rmf[rb][r], cb1, s[rb][1].f[r]);
      }
      #pragma unroll
      for (int r = 0; r < 8; r++) {
        float mx = redmax16(fmaxf(s[rb][0].f[r], s[rb][1].f[r]));
        float mn = fmaxf(mrow[rb][r], mx);
        float alpha = __expf(mrow[rb][r] - mn);
        float e0 = __expf(s[rb][0].f[r] - mn);
        float e1 = __expf(s[rb][1].f[r] - mn);
        float sm = redsum16(e0 + e1);
        lrow[rb][r] = lrow[rb][r] * alpha + sm;
        mrow[rb][r] = mn;
        s[rb][0].f[r] = e0; s[rb][1].f[r] = e1;
        o[rb][0].f[r] *= alpha; o[rb][1].f[r] *= alpha;
        o[rb][2].f[r] *= alpha; o[rb][3].f[r] *= alpha;
      }
      // P bounce: packed b32 stores (interleaved key order)
      #pragma unroll
      for (int r = 0; r < 8; r++) {
        unsigned pk = (unsigned)f2bf(s[rb][0].f[r]) |
                      ((unsigned)f2bf(s[rb][1].f[r]) << 16);
        *(unsigned*)&Pb[w][hsel * 8 + r][col * 2] = pk;
      }
      asm volatile("s_wait_dscnt 0x0" ::: "memory");  // wave-local LDS fence
      FragAB pa;
      pa.q[0] = *(const u32x4*)&Pb[w][col][hsel * 16];
      pa.q[1] = *(const u32x4*)&Pb[w][col][hsel * 16 + 8];
      #pragma unroll
      for (int tt = 0; tt < 4; tt++)
        o[rb][tt].v = wmma_bf16(pa.v, vf[tt].v, o[rb][tt].v);
    }

    // ---- commit next V tile, drain K copy, flip buffers ----
    if (more) {
      const unsigned short* ve = (const unsigned short*)&vvn;
      #pragma unroll
      for (int e = 0; e < 8; e++) VtT[nxt][c8 + e][kprm] = ve[e];
#if HAVE_TDM
      if (w == 0) __builtin_amdgcn_s_wait_tensorcnt(0);
#else
      wait_async0();
#endif
    }
    __syncthreads();
  }

  // ---- normalize and store bf16 head output ----
  #pragma unroll
  for (int rb = 0; rb < 2; rb++) {
    #pragma unroll
    for (int tt = 0; tt < 4; tt++) {
      #pragma unroll
      for (int r = 0; r < 8; r++) {
        int rr = qbase + rb * 16 + hsel * 8 + r;
        aob[(size_t)rr * INNER + h * DHEAD + tt * 16 + col] =
            f2bf(o[rb][tt].f[r] / lrow[rb][r]);
      }
    }
  }
}

// ---------------- launcher ----------------
extern "C" void kernel_launch(void* const* d_in, const int* in_sizes, int n_in,
                              void* d_out, int out_size, void* d_ws, size_t ws_size,
                              hipStream_t stream) {
  (void)in_sizes; (void)n_in; (void)out_size; (void)ws_size;
  const float* x     = (const float*)d_in[0];
  const float* ctx   = (const float*)d_in[1];
  const float* mask1 = (const float*)d_in[2];
  const float* mask2 = (const float*)d_in[3];
  const float* Wq    = (const float*)d_in[4];
  const float* Wk    = (const float*)d_in[5];
  const float* Wv    = (const float*)d_in[6];
  const float* Wo    = (const float*)d_in[7];
  const float* bo    = (const float*)d_in[8];
  float* out = (float*)d_out;

  char* wp = (char*)d_ws;
  unsigned short* qbuf = (unsigned short*)wp; wp += (size_t)NQ * INNER * 2;
  unsigned short* kbuf = (unsigned short*)wp; wp += (size_t)NK * INNER * 2;
  unsigned short* vbuf = (unsigned short*)wp; wp += (size_t)NK * INNER * 2;
  unsigned short* aob  = (unsigned short*)wp; wp += (size_t)NQ * INNER * 2;
  unsigned char* m1q   = (unsigned char*)wp;  wp += 8192;
  unsigned* m2kbits    = (unsigned*)wp;

  build_masks<<<dim3((NQ + 255) / 256), 256, 0, stream>>>(mask1, mask2, m1q, m2kbits);

  // Q projection carries the 1/sqrt(d) scale.
  gemm_wmma<false, false><<<dim3(NQ / 128, INNER / 64), 256, 0, stream>>>(
      x, Wq, nullptr, qbuf, NQ, INNER, QD, 0.125f);
  gemm_wmma<false, false><<<dim3(NK / 128, INNER / 64), 256, 0, stream>>>(
      ctx, Wk, nullptr, kbuf, NK, INNER, QD, 1.0f);
  gemm_wmma<false, false><<<dim3(NK / 128, INNER / 64), 256, 0, stream>>>(
      ctx, Wv, nullptr, vbuf, NK, INNER, QD, 1.0f);

  flash_attn<<<dim3(NQ / 256, HEADS), 256, 0, stream>>>(
      qbuf, kbuf, vbuf, m1q, m2kbits, aob);

  gemm_wmma<true, true><<<dim3(NQ / 128, QD / 64), 256, 0, stream>>>(
      aob, Wo, bo, out, NQ, QD, INNER, 1.0f);
}
